// entropy_BRM_45810121179228
// MI455X (gfx1250) — matrix operand order repaired and checked by reading, not compiled, not run on hardware
//
#include <hip/hip_runtime.h>
#include <hip/hip_bf16.h>
#include <math.h>

typedef __attribute__((ext_vector_type(16))) _Float16 v16h;
typedef __attribute__((ext_vector_type(8)))  _Float16 v8h;
typedef __attribute__((ext_vector_type(8)))  float    v8f;

#define NROWS 8192
#define DIM   768

// ---------------------------------------------------------------------------
// init row/col max buffers to -inf
// ---------------------------------------------------------------------------
__global__ void init_max_kernel(float* __restrict__ rowMax, float* __restrict__ colMax) {
    int i = blockIdx.x * blockDim.x + threadIdx.x;
    if (i < NROWS) {
        rowMax[i] = -INFINITY;
        colMax[i] = -INFINITY;
    }
}

// ---------------------------------------------------------------------------
// per-row L2 normalize fp32 -> fp16   (one block per row, 256 thr, 3 elem/thr)
// ---------------------------------------------------------------------------
__global__ __launch_bounds__(256) void normalize_kernel(const float* __restrict__ x,
                                                        _Float16* __restrict__ xh) {
    __shared__ float red[8];
    __shared__ float invs;
    const int row = blockIdx.x;
    const int t   = threadIdx.x;
    const float* p = x + (size_t)row * DIM;

    float v0 = p[t];
    float v1 = p[t + 256];
    float v2 = p[t + 512];
    float ss = v0 * v0 + v1 * v1 + v2 * v2;

    // wave32 reduction
    for (int off = 16; off; off >>= 1) ss += __shfl_xor(ss, off, 32);
    if ((t & 31) == 0) red[t >> 5] = ss;
    __syncthreads();
    if (t == 0) {
        float s = 0.f;
        for (int i = 0; i < 8; ++i) s += red[i];
        float n = fmaxf(sqrtf(s), 1e-8f);   // eps clamp as in reference
        invs = 1.0f / n;
    }
    __syncthreads();
    const float inv = invs;

    _Float16* q = xh + (size_t)row * DIM;
    q[t]       = (_Float16)(v0 * inv);
    q[t + 256] = (_Float16)(v1 * inv);
    q[t + 512] = (_Float16)(v2 * inv);
}

// ---------------------------------------------------------------------------
// sign-aware float atomic max via integer atomics (valid for any float order)
// ---------------------------------------------------------------------------
__device__ inline void atomicMaxF(float* addr, float v) {
    if (v >= 0.0f) atomicMax((int*)addr, __float_as_int(v));
    else           atomicMin((unsigned int*)addr, __float_as_uint(v));
}

// ---------------------------------------------------------------------------
// fused GEMM + row/col max.
//   block = 128 threads = 4 waves; each wave computes a 64x64 tile of C
//   using 4x4 accumulators of v_wmma_f32_16x16x32_f16 over K=768.
//   C[m][n] = sum_k X[m][k] * Y[n][k]   (both row-major fp16, normalized)
// ---------------------------------------------------------------------------
__global__ __launch_bounds__(128) void cosmax_gemm_kernel(const _Float16* __restrict__ X,
                                                          const _Float16* __restrict__ Y,
                                                          float* __restrict__ rowMax,
                                                          float* __restrict__ colMax) {
    const int lane  = threadIdx.x & 31;
    const int wave  = threadIdx.x >> 5;
    const int mBase = blockIdx.y * 128 + (wave >> 1) * 64;
    const int nBase = blockIdx.x * 128 + (wave & 1) * 64;
    const int r16   = lane & 15;   // row within 16-row fragment (A), col (B/C)
    const int half  = lane >> 4;   // lane half selects K interleave

    v8f acc[4][4] = {};

    // A fragment lane base: contiguous K order used here -> lane holds K {0..15} or {16..31}
    const _Float16* pa0 = X + (size_t)(mBase + r16) * DIM + half * 16;
    // B fragment lane base: same contiguous K order as A -> lane holds col N=r16, K {0..15} or {16..31}
    const _Float16* pb0 = Y + (size_t)(nBase + r16) * DIM + half * 16;

    for (int k = 0; k < DIM; k += 32) {
        v16h a[4], b[4];
#pragma unroll
        for (int i = 0; i < 4; ++i) {
            const _Float16* pa = pa0 + (size_t)(16 * i) * DIM + k;
            v8h lo = *(const v8h*)pa;          // K = k + half*16 + 0..7
            v8h hi = *(const v8h*)(pa + 8);    // K = k + half*16 + 8..15
            a[i] = __builtin_shufflevector(lo, hi, 0,1,2,3,4,5,6,7,8,9,10,11,12,13,14,15);
            const _Float16* pb = pb0 + (size_t)(16 * i) * DIM + k;
            b[i] = *(const v16h*)pb;           // K = k + half*16 + 0..15
        }
#pragma unroll
        for (int i = 0; i < 4; ++i)
#pragma unroll
            for (int j = 0; j < 4; ++j)
                acc[i][j] = __builtin_amdgcn_wmma_f32_16x16x32_f16(
                    false, a[i], false, b[j], (short)0, acc[i][j], false, false);
    }

    // C/D layout: acc[i][j][r] on lane L = C[16i + r + 8*(L>>4)][16j + (L&15)]
    // --- row max over this tile's 64 columns -> atomic merge ---
#pragma unroll
    for (int i = 0; i < 4; ++i) {
#pragma unroll
        for (int r = 0; r < 8; ++r) {
            float m = fmaxf(fmaxf(acc[i][0][r], acc[i][1][r]),
                            fmaxf(acc[i][2][r], acc[i][3][r]));
            for (int off = 8; off; off >>= 1) m = fmaxf(m, __shfl_xor(m, off, 32));
            if (r16 == 0)
                atomicMaxF(&rowMax[mBase + 16 * i + r + 8 * half], m);
        }
    }

    // --- col max over this tile's 64 rows -> atomic merge ---
#pragma unroll
    for (int j = 0; j < 4; ++j) {
        float m = -INFINITY;
#pragma unroll
        for (int i = 0; i < 4; ++i)
#pragma unroll
            for (int r = 0; r < 8; ++r)
                m = fmaxf(m, acc[i][j][r]);
        m = fmaxf(m, __shfl_xor(m, 16, 32));
        if (half == 0)
            atomicMaxF(&colMax[nBase + 16 * j + r16], m);
    }
}

// ---------------------------------------------------------------------------
// entropy terms: out[0] = -sum(exp(lp(rowMax))*lp(rowMax)), out[1] likewise cols
// single block -> deterministic, no atomics on d_out
// ---------------------------------------------------------------------------
__global__ __launch_bounds__(256) void finalize_kernel(const float* __restrict__ rowMax,
                                                       const float* __restrict__ colMax,
                                                       float* __restrict__ out) {
    __shared__ float red[8];
    const float inv2s2   = 1.0f / (2.0f * 0.8f * 0.8f);   // sigma = 0.8
    const float log_norm = -0.6957949819f;                // -ln(0.8) - 0.5*ln(2*pi)
    const int t = threadIdx.x;
    for (int which = 0; which < 2; ++which) {
        const float* v = which ? colMax : rowMax;
        float s = 0.f;
        for (int i = t; i < NROWS; i += 256) {
            float c  = v[i];
            float lp = -(c * c) * inv2s2 + log_norm;
            s += -expf(lp) * lp;
        }
        for (int off = 16; off; off >>= 1) s += __shfl_xor(s, off, 32);
        if ((t & 31) == 0) red[t >> 5] = s;
        __syncthreads();
        if (t == 0) {
            float tot = 0.f;
            for (int i = 0; i < 8; ++i) tot += red[i];
            out[which] = tot;
        }
        __syncthreads();
    }
}

// ---------------------------------------------------------------------------
extern "C" void kernel_launch(void* const* d_in, const int* in_sizes, int n_in,
                              void* d_out, int out_size, void* d_ws, size_t ws_size,
                              hipStream_t stream) {
    const float* ex = (const float*)d_in[0];
    const float* ey = (const float*)d_in[1];
    float* out = (float*)d_out;

    char* ws = (char*)d_ws;
    const size_t matBytes = (size_t)NROWS * DIM * sizeof(_Float16);  // 12.6 MB each
    _Float16* exh   = (_Float16*)(ws);
    _Float16* eyh   = (_Float16*)(ws + matBytes);
    float*    rowMx = (float*)(ws + 2 * matBytes);
    float*    colMx = (float*)(ws + 2 * matBytes + (size_t)NROWS * sizeof(float));

    init_max_kernel<<<(NROWS + 255) / 256, 256, 0, stream>>>(rowMx, colMx);
    normalize_kernel<<<NROWS, 256, 0, stream>>>(ex, exh);
    normalize_kernel<<<NROWS, 256, 0, stream>>>(ey, eyh);

    dim3 grid(NROWS / 128, NROWS / 128);   // 64 x 64 blocks, 4 waves each
    cosmax_gemm_kernel<<<grid, 128, 0, stream>>>(exh, eyh, rowMx, colMx);

    finalize_kernel<<<1, 256, 0, stream>>>(rowMx, colMx, out);
}
